// JinaEmbeddingsV3SelfOutput_74809740362143
// MI455X (gfx1250) — matrix operand order repaired and checked by reading, not compiled, not run on hardware
//
#include <hip/hip_runtime.h>

// ---------------------------------------------------------------------------
// JinaEmbeddingsV3SelfOutput fused pipeline for MI455X (gfx1250, wave32).
//   h1    = x @ W^T + b
//   base2 = h1 @ W^T + b
//   low   = h1 @ A_task^T            (rank 4)
//   h2    = base2 + 0.25 * low @ B_task^T
//   out   = LayerNorm(h2 + input) * gamma + beta
// GEMMs: v_wmma_f32_16x16x32_bf16, operands staged in LDS via
// GLOBAL_LOAD_ASYNC_TO_LDS_B128 (ASYNCcnt, double buffered); B fragments
// additionally software-pipelined two-deep out of LDS so ds_load latency
// overlaps the XDL WMMA ops instead of serializing (s_wait_dscnt 0x0).
// ---------------------------------------------------------------------------

#define D        1024
#define MTOT     32768       // 16 * 2048 rows
#define SEQ      2048
#define RANKV    4
#define MBLK     32          // rows per workgroup in GEMM kernels
#define LN_EPS   1e-5f
#define LORA_SC  0.25f       // 1/rank

#define BPADC    1032        // 1024 + 8 bf16 pad  (row stride 2064B, 16B mult)
#define APADC2   40          // 32 + 8 bf16 pad    (80B, 16B mult)
#define APADC1   36          // 32 + 4 f32 pad     (144B, 16B mult)

typedef __attribute__((ext_vector_type(16))) __bf16 v16bf;
typedef __attribute__((ext_vector_type(8)))  float  v8f;
typedef int v4i __attribute__((vector_size(4 * sizeof(int))));

union BF16Frag {
  v16bf v;
  unsigned short u[16];
  unsigned int   w[8];
  uint4 q[2];
};

__device__ __forceinline__ unsigned short f2bf(float f) {
  unsigned int u = __float_as_uint(f);
  u += 0x7FFFu + ((u >> 16) & 1u);          // round-to-nearest-even
  return (unsigned short)(u >> 16);
}
__device__ __forceinline__ float bf2f(unsigned short h) {
  return __uint_as_float(((unsigned int)h) << 16);
}

#if __has_builtin(__builtin_amdgcn_cvt_pk_bf16_f32)
__device__ __forceinline__ unsigned int pk_bf16(float x, float y) {
  auto r = __builtin_amdgcn_cvt_pk_bf16_f32(x, y);   // v_cvt_pk_bf16_f32
  return __builtin_bit_cast(unsigned int, r);
}
#else
__device__ __forceinline__ unsigned int pk_bf16(float x, float y) {
  return (unsigned int)f2bf(x) | ((unsigned int)f2bf(y) << 16);
}
#endif

// ---- CDNA5 async global->LDS copy (ASYNCcnt) ------------------------------
#if __has_builtin(__builtin_amdgcn_global_load_async_to_lds_b128)
#define USE_ASYNC 1
#else
#define USE_ASYNC 0
#endif

__device__ __forceinline__ void async_b128(void* lds, const void* g) {
#if USE_ASYNC
  __builtin_amdgcn_global_load_async_to_lds_b128(
      (__attribute__((address_space(1))) v4i*)g,
      (__attribute__((address_space(3))) v4i*)lds, 0, 0);
#else
  *(uint4*)lds = *(const uint4*)g;
#endif
}
__device__ __forceinline__ void wait_async() {
#if USE_ASYNC
#if __has_builtin(__builtin_amdgcn_s_wait_asynccnt)
  __builtin_amdgcn_s_wait_asynccnt(0);
#else
  asm volatile("s_wait_asynccnt 0x0" ::: "memory");
#endif
#endif
}

// ---------------------------------------------------------------------------
// W[e][d] fp32 -> Wt[d][e] bf16  (LDS-tiled transpose; Wt is 2MB, L2-resident)
// ---------------------------------------------------------------------------
__global__ void k_transpose_w(const float* __restrict__ W,
                              unsigned short* __restrict__ Wt) {
  __shared__ float tile[32][33];
  const int tx = threadIdx.x, ty = threadIdx.y;       // blockDim = (32, 8)
  const int db = blockIdx.x * 32, eb = blockIdx.y * 32;
#pragma unroll
  for (int i = 0; i < 4; ++i)
    tile[ty + i * 8][tx] = W[(eb + ty + i * 8) * D + (db + tx)];
  __syncthreads();
#pragma unroll
  for (int i = 0; i < 4; ++i)
    Wt[(db + ty + i * 8) * D + (eb + tx)] = f2bf(tile[tx][ty + i * 8]);
}

// ---------------------------------------------------------------------------
// GEMM1: H1(bf16) = bf16( X(fp32) @ Wt + bias )
// Block: 256 threads = 8 waves; 32 rows x 1024 cols per block.
// K-chunks of 32 staged in LDS (double buffered, async copies):
//   sB[buf][32][1032] bf16 : Wt rows kc..kc+31, all 1024 cols (64KB/buf)
//   sAf[buf][32][36]  f32  : X rows of the block, cols kc..kc+31
// A-frag (16-bit A 16x32): lane<16 -> K {0..7,16..23}; lane>=16 -> +8.
// B-frag (32x16): lane%16 = N col; K {0..7,16..23}, lane>=16 -> +8.
// ---------------------------------------------------------------------------
__global__ void __launch_bounds__(256) k_gemm1(
    const float* __restrict__ X, const unsigned short* __restrict__ Wt,
    const float* __restrict__ bias, unsigned short* __restrict__ H1) {
  __shared__ unsigned short sB[2][32][BPADC];
  __shared__ float sAf[2][32][APADC1];

  const int tid  = threadIdx.x;
  const int lane = tid & 31;
  const int wid  = tid >> 5;
  const int wm   = wid >> 2;          // 0..1
  const int we   = wid & 3;           // 0..3
  const int lm   = lane & 15;
  const bool hi  = lane >= 16;
  const int row0 = blockIdx.x * MBLK + wm * 16;
  const int e0   = we * 256;

  // cooperative chunk loader: B 64KB (16 b128/thread) + A 4KB (1 b128/thread)
  const int lr = tid >> 3;            // 0..31 (row in chunk)
  const int lc = tid & 7;             // 8 threads per row
  auto load_chunk = [&](int kc, int bb) {
    const unsigned short* gB = Wt + (kc + lr) * D + lc * 128;
    unsigned short* lB = &sB[bb][lr][lc * 128];
#pragma unroll
    for (int i = 0; i < 16; ++i)
      async_b128(lB + i * 8, gB + i * 8);
    async_b128(&sAf[bb][lr][lc * 4],
               X + (blockIdx.x * MBLK + lr) * D + kc + lc * 4);
  };

  v8f acc[16];
#pragma unroll
  for (int t = 0; t < 16; ++t)
#pragma unroll
    for (int j = 0; j < 8; ++j) acc[t][j] = 0.0f;

  load_chunk(0, 0);
  for (int kc = 0; kc < D; kc += 32) {
    const int cur = (kc >> 5) & 1;
    wait_async();
    __syncthreads();
    if (kc + 32 < D) load_chunk(kc + 32, cur ^ 1);

    // A fragment from LDS fp32, packed-convert to bf16
    BF16Frag A;
    const float* ap = &sAf[cur][lm][hi ? 8 : 0];
    const float4 a0 = *(const float4*)ap;
    const float4 a1 = *(const float4*)(ap + 4);
    const float4 a2 = *(const float4*)(ap + 16);
    const float4 a3 = *(const float4*)(ap + 20);
    A.w[0] = pk_bf16(a0.x, a0.y); A.w[1] = pk_bf16(a0.z, a0.w);
    A.w[2] = pk_bf16(a1.x, a1.y); A.w[3] = pk_bf16(a1.z, a1.w);
    A.w[4] = pk_bf16(a2.x, a2.y); A.w[5] = pk_bf16(a2.z, a2.w);
    A.w[6] = pk_bf16(a3.x, a3.y); A.w[7] = pk_bf16(a3.z, a3.w);

    // B fragments: two-deep register pipeline over LDS loads so each WMMA
    // overlaps with the next tile's 16 strided 16-bit LDS loads.
    const unsigned short* bp = &sB[cur][hi ? 8 : 0][e0 + lm];
    BF16Frag Bf[2];
#pragma unroll
    for (int s = 0; s < 16; ++s)
      Bf[0].u[s] = bp[(s < 8 ? s : s + 8) * BPADC];
#pragma unroll
    for (int t = 0; t < 16; ++t) {
      if (t + 1 < 16) {
#pragma unroll
        for (int s = 0; s < 16; ++s)
          Bf[(t + 1) & 1].u[s] = bp[(s < 8 ? s : s + 8) * BPADC + (t + 1) * 16];
      }
      acc[t] = __builtin_amdgcn_wmma_f32_16x16x32_bf16(
          false, A.v, false, Bf[t & 1].v, (short)0, acc[t], false, false);
    }
  }

  // Epilogue: bias add, write bf16. C layout: lane<16 -> N=lane, M=v;
  // lane>=16 -> N=lane-16, M=8+v.
#pragma unroll
  for (int t = 0; t < 16; ++t) {
    const int e = e0 + t * 16 + lm;
    const float be = bias[e];
#pragma unroll
    for (int v = 0; v < 8; ++v) {
      const int gr = row0 + (hi ? 8 + v : v);
      H1[gr * D + e] = f2bf(acc[t][v] + be);
    }
  }
}

// ---------------------------------------------------------------------------
// low[row][r] = sum_d H1[row][d] * loraA[task][r][d]   (rank 4, wave-per-row)
// ---------------------------------------------------------------------------
__global__ void __launch_bounds__(256) k_low(
    const unsigned short* __restrict__ H1, const float* __restrict__ loraA,
    const int* __restrict__ mask, float* __restrict__ low) {
  const int lane = threadIdx.x & 31;
  const int wid  = threadIdx.x >> 5;
  const int row  = blockIdx.x * 8 + wid;
  const int task = mask[row / SEQ];
  const float* Ab = loraA + task * (RANKV * D);
  const unsigned short* hp = H1 + row * D + lane * 32;

  float a0 = 0.f, a1 = 0.f, a2 = 0.f, a3 = 0.f;
#pragma unroll
  for (int c = 0; c < 4; ++c) {
    union { uint4 q; unsigned short u[8]; } xx;
    xx.q = ((const uint4*)(hp + c * 8))[0];
#pragma unroll
    for (int j = 0; j < 8; ++j) {
      const float xv = bf2f(xx.u[j]);
      const int k = lane * 32 + c * 8 + j;
      a0 += xv * Ab[0 * D + k];
      a1 += xv * Ab[1 * D + k];
      a2 += xv * Ab[2 * D + k];
      a3 += xv * Ab[3 * D + k];
    }
  }
#pragma unroll
  for (int off = 16; off > 0; off >>= 1) {
    a0 += __shfl_xor(a0, off, 32);
    a1 += __shfl_xor(a1, off, 32);
    a2 += __shfl_xor(a2, off, 32);
    a3 += __shfl_xor(a3, off, 32);
  }
  if (lane == 0) ((float4*)(low + row * 4))[0] = make_float4(a0, a1, a2, a3);
}

// ---------------------------------------------------------------------------
// GEMM2 fused: out = LN( H1 @ Wt + b + 0.25 * low @ B_task^T + resid )
// Same tiling; A chunk (bf16) also async-staged. Each block owns a full
// 1024-wide row slab so LayerNorm reduces via LDS ds_add_f32 atomics.
// ---------------------------------------------------------------------------
__global__ void __launch_bounds__(256) k_gemm2(
    const unsigned short* __restrict__ H1, const unsigned short* __restrict__ Wt,
    const float* __restrict__ bias, const float* __restrict__ low,
    const float* __restrict__ loraB, const int* __restrict__ mask,
    const float* __restrict__ resid, const float* __restrict__ gamma,
    const float* __restrict__ beta, float* __restrict__ out) {
  __shared__ unsigned short sB[2][32][BPADC];
  __shared__ unsigned short sA[2][32][APADC2];
  __shared__ float s_sum[MBLK];
  __shared__ float s_sq[MBLK];

  const int tid  = threadIdx.x;
  const int lane = tid & 31;
  const int wid  = tid >> 5;
  const int wm   = wid >> 2;
  const int we   = wid & 3;
  const int lm   = lane & 15;
  const bool hi  = lane >= 16;
  const int row0 = blockIdx.x * MBLK + wm * 16;
  const int e0   = we * 256;

  if (tid < MBLK) { s_sum[tid] = 0.f; s_sq[tid] = 0.f; }

  const int lr = tid >> 3;
  const int lc = tid & 7;
  auto load_chunk = [&](int kc, int bb) {
    const unsigned short* gB = Wt + (kc + lr) * D + lc * 128;
    unsigned short* lB = &sB[bb][lr][lc * 128];
#pragma unroll
    for (int i = 0; i < 16; ++i)
      async_b128(lB + i * 8, gB + i * 8);
    if (tid < 128) {  // A chunk: 32 rows x 64B, 4 threads/row
      const int r = tid >> 2, c = (tid & 3) * 8;
      async_b128(&sA[bb][r][c],
                 H1 + (blockIdx.x * MBLK + r) * D + kc + c);
    }
  };

  v8f acc[16];
#pragma unroll
  for (int t = 0; t < 16; ++t)
#pragma unroll
    for (int j = 0; j < 8; ++j) acc[t][j] = 0.0f;

  load_chunk(0, 0);
  for (int kc = 0; kc < D; kc += 32) {
    const int cur = (kc >> 5) & 1;
    wait_async();
    __syncthreads();
    if (kc + 32 < D) load_chunk(kc + 32, cur ^ 1);

    BF16Frag A;
    const unsigned short* ap = &sA[cur][lm][hi ? 8 : 0];
    A.q[0] = *(const uint4*)ap;
    A.q[1] = *(const uint4*)(ap + 16);

    const unsigned short* bp = &sB[cur][hi ? 8 : 0][e0 + lm];
    BF16Frag Bf[2];
#pragma unroll
    for (int s = 0; s < 16; ++s)
      Bf[0].u[s] = bp[(s < 8 ? s : s + 8) * BPADC];
#pragma unroll
    for (int t = 0; t < 16; ++t) {
      if (t + 1 < 16) {
#pragma unroll
        for (int s = 0; s < 16; ++s)
          Bf[(t + 1) & 1].u[s] = bp[(s < 8 ? s : s + 8) * BPADC + (t + 1) * 16];
      }
      acc[t] = __builtin_amdgcn_wmma_f32_16x16x32_bf16(
          false, A.v, false, Bf[t & 1].v, (short)0, acc[t], false, false);
    }
  }

  // ---- Pass 1: bias + LoRA delta + residual; accumulate LN statistics.
  const int task = mask[(blockIdx.x * MBLK) / SEQ];
  float4 lw[8];
#pragma unroll
  for (int v = 0; v < 8; ++v) {
    const int gr = row0 + (hi ? 8 + v : v);
    lw[v] = ((const float4*)(low + gr * 4))[0];
  }
  float sums[8], sqs[8];
#pragma unroll
  for (int v = 0; v < 8; ++v) { sums[v] = 0.f; sqs[v] = 0.f; }

#pragma unroll
  for (int t = 0; t < 16; ++t) {
    const int e = e0 + t * 16 + lm;
    const float be = bias[e];
    const float4 Bw = ((const float4*)(loraB + (task * D + e) * RANKV))[0];
#pragma unroll
    for (int v = 0; v < 8; ++v) {
      const int gr = row0 + (hi ? 8 + v : v);
      float val = acc[t][v] + be
                + LORA_SC * (lw[v].x * Bw.x + lw[v].y * Bw.y +
                             lw[v].z * Bw.z + lw[v].w * Bw.w)
                + resid[gr * D + e];
      acc[t][v] = val;
      sums[v] += val;
      sqs[v]  += val * val;
    }
  }

#pragma unroll
  for (int v = 0; v < 8; ++v) {
    const int rl = wm * 16 + (hi ? 8 + v : v);
    atomicAdd(&s_sum[rl], sums[v]);   // ds_add_f32
    atomicAdd(&s_sq[rl],  sqs[v]);
  }
  __syncthreads();

  float mu[8], rs[8];
#pragma unroll
  for (int v = 0; v < 8; ++v) {
    const int rl = wm * 16 + (hi ? 8 + v : v);
    const float m   = s_sum[rl] * (1.0f / D);
    const float var = s_sq[rl] * (1.0f / D) - m * m;
    mu[v] = m;
    rs[v] = rsqrtf(var + LN_EPS);
  }

  // ---- Pass 2: normalize + affine + store.
#pragma unroll
  for (int t = 0; t < 16; ++t) {
    const int e = e0 + t * 16 + lm;
    const float g  = gamma[e];
    const float bb = beta[e];
#pragma unroll
    for (int v = 0; v < 8; ++v) {
      const int gr = row0 + (hi ? 8 + v : v);
      out[gr * D + e] = (acc[t][v] - mu[v]) * rs[v] * g + bb;
    }
  }
}

// ---------------------------------------------------------------------------
extern "C" void kernel_launch(void* const* d_in, const int* in_sizes, int n_in,
                              void* d_out, int out_size, void* d_ws, size_t ws_size,
                              hipStream_t stream) {
  (void)in_sizes; (void)n_in; (void)out_size; (void)ws_size;
  const float* hidden = (const float*)d_in[0];
  const float* input  = (const float*)d_in[1];
  const float* W      = (const float*)d_in[2];
  const float* b      = (const float*)d_in[3];
  const float* loraA  = (const float*)d_in[4];
  const float* loraB  = (const float*)d_in[5];
  const float* gamma  = (const float*)d_in[6];
  const float* beta   = (const float*)d_in[7];
  const int*   mask   = (const int*)d_in[8];
  float* out = (float*)d_out;

  char* ws = (char*)d_ws;
  unsigned short* Wt = (unsigned short*)ws;                              // 2 MB
  unsigned short* H1 = (unsigned short*)(ws + (2ull << 20));             // 64 MB
  float*          low = (float*)(ws + (2ull << 20) + (64ull << 20));     // 512 KB

  k_transpose_w<<<dim3(D / 32, D / 32), dim3(32, 8), 0, stream>>>(W, Wt);
  k_gemm1<<<MTOT / MBLK, 256, 0, stream>>>(hidden, Wt, b, H1);
  k_low<<<MTOT / 8, 256, 0, stream>>>(H1, loraA, mask, low);
  k_gemm2<<<MTOT / MBLK, 256, 0, stream>>>(H1, Wt, b, low, loraB, mask,
                                           input, gamma, beta, out);
}
